// Encoder_41128606826552
// MI455X (gfx1250) — compile-verified, matrix operand order checked by reading, not run on hardware
//
#include <hip/hip_runtime.h>
#include <hip/hip_bf16.h>
#include <cstdint>
#include <cstddef>

// ---------------------------------------------------------------------------
// PointNet++ MSG encoder for MI455X (gfx1250, wave32, WMMA bf16)
// Weights pre-swizzled into per-lane WMMA fragment order -> vector loads.
// ---------------------------------------------------------------------------

typedef __attribute__((ext_vector_type(8)))  unsigned short v8u;
typedef __attribute__((ext_vector_type(16))) unsigned short v16u;
typedef __attribute__((ext_vector_type(16))) __bf16         v16bf;
typedef __attribute__((ext_vector_type(8)))  float          v8f;

struct LayerP {
    const unsigned short* W;   // bf16, fragment-swizzled: [(kc*ntiles+nt)*512 + lane*16 + j]
    const float* s;            // [cout] BN scale
    const float* b;            // [cout] BN shift
    int cin, cinPad, cout;
};
struct MlpP { LayerP L[3]; };

__device__ __forceinline__ unsigned short f2bf(float f) {
    unsigned int u = __builtin_bit_cast(unsigned int, f);
    u += 0x7FFFu + ((u >> 16) & 1u);           // round-to-nearest-even
    return (unsigned short)(u >> 16);
}

// ---------------------------------------------------------------------------
// A fragment: 16x32 bf16 tile from LDS activation [rows, stride] at row-tile mt,
// K-chunk kc. ISA 7.12.2 layout (16-bit A 16x32):
//   lanes 0-15: v[0..7]=K 0..7,  v[8..15]=K 16..23   (M = lane)
//   lanes16-31: v[0..7]=K 8..15, v[8..15]=K 24..31   (M = lane-16)
// Both 8-half runs are contiguous & 16B aligned -> two ds_load_b128.
// ---------------------------------------------------------------------------
__device__ __forceinline__ v16bf load_a_frag(const unsigned short* act, int stride,
                                             int mt, int kc, int lane) {
    int row = mt * 16 + (lane & 15);
    int kb  = (lane >> 4) ? 8 : 0;
    const unsigned short* p = act + row * stride + kc * 32 + kb;
    v8u lo = *(const v8u*)p;
    v8u hi = *(const v8u*)(p + 16);
    v16u v;
#pragma unroll
    for (int j = 0; j < 8; ++j) { v[j] = lo[j]; v[8 + j] = hi[j]; }
    return __builtin_bit_cast(v16bf, v);
}

// B fragment from pre-swizzled weights: lane reads its 16 halves contiguously
// (32 bytes, 32B aligned -> two global_load_b128).
__device__ __forceinline__ v16bf load_b_frag(const unsigned short* Wsw, int ntiles,
                                             int nt, int kc, int lane) {
    const v16u* p = (const v16u*)(Wsw + (((size_t)kc * ntiles + nt) << 9) + (lane << 4));
    return __builtin_bit_cast(v16bf, *p);
}

// ---------------------------------------------------------------------------
// Split input [B,6,N] -> pts [B,N,3], nrm [B,N,3]
// ---------------------------------------------------------------------------
__global__ void split_xyz_kernel(const float* __restrict__ in,
                                 float* __restrict__ pts, float* __restrict__ nrm,
                                 int B, int N) {
    int t = blockIdx.x * blockDim.x + threadIdx.x;
    if (t >= B * N) return;
    int b = t / N, n = t - b * N;
#pragma unroll
    for (int c = 0; c < 3; ++c) {
        pts[(size_t)t * 3 + c] = in[((size_t)b * 6 + c) * N + n];
        nrm[(size_t)t * 3 + c] = in[((size_t)b * 6 + 3 + c) * N + n];
    }
}

// Convert weight f32 [cout,cin] -> bf16 fragment-swizzled order, zero-padded K.
// Element t: tile = t>>9 (kc,nt), r = t&511 -> lane = r>>4, j = r&15.
__global__ void convw_kernel(const float* __restrict__ W, unsigned short* __restrict__ Wsw,
                             int cin, int cinPad, int cout) {
    int t = blockIdx.x * blockDim.x + threadIdx.x;
    int ntiles = cout >> 4;
    int tot = (cinPad >> 5) * ntiles * 512;
    if (t >= tot) return;
    int tile = t >> 9, r = t & 511;
    int lane = r >> 4, j = r & 15;
    int kc = tile / ntiles, nt = tile - kc * ntiles;
    int n  = nt * 16 + (lane & 15);
    int kb = (lane >> 4) ? 8 : 0;
    int k  = kc * 32 + ((j < 8) ? (kb + j) : (16 + kb + (j - 8)));
    float v = (k < cin) ? W[(size_t)n * cin + k] : 0.0f;
    Wsw[t] = f2bf(v);
}

// ---------------------------------------------------------------------------
// Farthest point sampling, one block per batch. Deterministic start idx 0,
// first-occurrence argmax (matches jnp.argmax).
// ---------------------------------------------------------------------------
#define FPS_THREADS 256
template <int NPTS>
__global__ void fps_kernel(const float* __restrict__ pts, int npoint,
                           int* __restrict__ idx_out, float* __restrict__ new_xyz) {
    __shared__ float sx[NPTS], sy[NPTS], sz[NPTS], sdist[NPTS];
    __shared__ float rv[FPS_THREADS];
    __shared__ int   ri[FPS_THREADS];
    __shared__ int   sfar;
    int b = blockIdx.x, tid = threadIdx.x;
    for (int n = tid; n < NPTS; n += FPS_THREADS) {
        size_t base = ((size_t)b * NPTS + n) * 3;
        sx[n] = pts[base + 0]; sy[n] = pts[base + 1]; sz[n] = pts[base + 2];
        sdist[n] = 1e10f;
    }
    if (tid == 0) sfar = 0;
    __syncthreads();
    for (int it = 0; it < npoint; ++it) {
        int far = sfar;
        if (tid == 0) {
            idx_out[b * npoint + it] = far;
            size_t o = ((size_t)b * npoint + it) * 3;
            new_xyz[o + 0] = sx[far]; new_xyz[o + 1] = sy[far]; new_xyz[o + 2] = sz[far];
        }
        float cx = sx[far], cy = sy[far], cz = sz[far];
        float bv = -1.0f; int bi = 0;
        for (int n = tid; n < NPTS; n += FPS_THREADS) {
            float dx = sx[n] - cx, dy = sy[n] - cy, dz = sz[n] - cz;
            float d  = dx * dx + dy * dy + dz * dz;
            float nd = fminf(sdist[n], d);
            sdist[n] = nd;
            if (nd > bv) { bv = nd; bi = n; }
        }
        rv[tid] = bv; ri[tid] = bi;
        __syncthreads();
        for (int off = FPS_THREADS >> 1; off > 0; off >>= 1) {
            if (tid < off) {
                if (rv[tid + off] > rv[tid] ||
                    (rv[tid + off] == rv[tid] && ri[tid + off] < ri[tid])) {
                    rv[tid] = rv[tid + off]; ri[tid] = ri[tid + off];
                }
            }
            __syncthreads();
        }
        if (tid == 0) sfar = ri[0];
        __syncthreads();
    }
}

// ---------------------------------------------------------------------------
// Ball query: first K in-radius indices in ascending order; pad with first hit
// ---------------------------------------------------------------------------
__global__ void ballquery_kernel(const float* __restrict__ pts, int Npts,
                                 const float* __restrict__ new_xyz, int S, int K,
                                 float r2, int* __restrict__ gidx, int B) {
    int t = blockIdx.x * blockDim.x + threadIdx.x;
    if (t >= B * S) return;
    int b = t / S;
    float cx = new_xyz[(size_t)t * 3 + 0];
    float cy = new_xyz[(size_t)t * 3 + 1];
    float cz = new_xyz[(size_t)t * 3 + 2];
    int* out = gidx + (size_t)t * K;
    int cnt = 0, first = -1;
    for (int n = 0; n < Npts && cnt < K; ++n) {
        size_t base = ((size_t)b * Npts + n) * 3;
        float dx = pts[base + 0] - cx, dy = pts[base + 1] - cy, dz = pts[base + 2] - cz;
        float d2 = dx * dx + dy * dy + dz * dz;
        if (d2 <= r2) { if (first < 0) first = n; out[cnt++] = n; }
    }
    if (first < 0) first = 0;
    for (; cnt < K; ++cnt) out[cnt] = first;
}

// ---------------------------------------------------------------------------
// Fused grouped MLP + max-pool. One block (4 waves) per (b,s) group.
//  mode 0: features = [feats[gidx], xyz[gidx]-center]   (SA stages)
//  mode 1: features = [coords[r], feats[r]]             (group_all, SA3)
// Layers ping-pong bufA -> bufB -> bufA; final layer fuses max over K.
// ---------------------------------------------------------------------------
__global__ __launch_bounds__(128)
void sa_mlp_kernel(const float* __restrict__ coords, int Npts,
                   const float* __restrict__ feats, int Cf,
                   const float* __restrict__ new_xyz, const int* __restrict__ gidx,
                   int S, int K, int mode,
                   float* __restrict__ outp, int outStride, int outOff,
                   MlpP P, int bufAElems) {
    extern __shared__ unsigned short smem_us[];
    unsigned short* bufA = smem_us;
    unsigned short* bufB = smem_us + bufAElems;

    int b = blockIdx.x / S, s = blockIdx.x - b * S;
    int tid = threadIdx.x, lane = tid & 31, wave = tid >> 5;
    const int NW = blockDim.x >> 5;

    // ---- gather grouped input into bufA (bf16, zero-padded cols) ----
    int cinPad0 = P.L[0].cinPad;
    for (int e = tid; e < K * cinPad0; e += blockDim.x) {
        int r = e / cinPad0, c = e - r * cinPad0;
        float v = 0.0f;
        if (mode == 0) {
            int idx = gidx[((size_t)(b * S + s)) * K + r];
            if (c < Cf)
                v = feats[((size_t)b * Npts + idx) * Cf + c];
            else if (c < Cf + 3)
                v = coords[((size_t)b * Npts + idx) * 3 + (c - Cf)]
                  - new_xyz[((size_t)(b * S + s)) * 3 + (c - Cf)];
        } else {
            if (c < 3)
                v = coords[((size_t)b * K + r) * 3 + c];
            else if (c < 3 + Cf)
                v = feats[((size_t)b * K + r) * Cf + (c - 3)];
        }
        bufA[r * cinPad0 + c] = f2bf(v);
    }
    __syncthreads();

    const unsigned short* in = bufA;
    unsigned short* outbuf = bufB;
    int inStride = cinPad0;
    int mtiles = K >> 4;

    for (int l = 0; l < 3; ++l) {
        LayerP Lp = P.L[l];
        int ntiles = Lp.cout >> 4;
        int kch    = Lp.cinPad >> 5;
        if (l < 2) {
            int jobs = mtiles * ntiles;
            for (int j = wave; j < jobs; j += NW) {
                int mt = j / ntiles, nt = j - mt * ntiles;
                v8f acc = {0.f, 0.f, 0.f, 0.f, 0.f, 0.f, 0.f, 0.f};
                for (int kc = 0; kc < kch; ++kc) {
                    if (kc + 1 < kch)
                        __builtin_prefetch(Lp.W + (((size_t)(kc + 1) * ntiles + nt) << 9), 0, 0);
                    v16bf a  = load_a_frag(in, inStride, mt, kc, lane);
                    v16bf bb = load_b_frag(Lp.W, ntiles, nt, kc, lane);
                    acc = __builtin_amdgcn_wmma_f32_16x16x32_bf16(
                              false, a, false, bb, (short)0, acc, false, false);
                }
                int   ncol = nt * 16 + (lane & 15);
                float sc = Lp.s[ncol], bi = Lp.b[ncol];
                int   rbase = mt * 16 + ((lane >> 4) << 3);
#pragma unroll
                for (int i = 0; i < 8; ++i) {
                    float v = acc[i] * sc + bi;
                    v = v > 0.0f ? v : 0.0f;
                    outbuf[(rbase + i) * Lp.cout + ncol] = f2bf(v);
                }
            }
            __syncthreads();
            in = outbuf;
            inStride = Lp.cout;
            outbuf = (outbuf == bufB) ? bufA : bufB;
        } else {
            // final layer: fuse max over K (rows), store pooled [cout] slice
            for (int nt = wave; nt < ntiles; nt += NW) {
                int   ncol = nt * 16 + (lane & 15);
                float sc = Lp.s[ncol], bi = Lp.b[ncol];
                float rmax[8];
#pragma unroll
                for (int i = 0; i < 8; ++i) rmax[i] = -1e30f;
                for (int mt = 0; mt < mtiles; ++mt) {
                    v8f acc = {0.f, 0.f, 0.f, 0.f, 0.f, 0.f, 0.f, 0.f};
                    for (int kc = 0; kc < kch; ++kc) {
                        if (kc + 1 < kch)
                            __builtin_prefetch(Lp.W + (((size_t)(kc + 1) * ntiles + nt) << 9), 0, 0);
                        v16bf a  = load_a_frag(in, inStride, mt, kc, lane);
                        v16bf bb = load_b_frag(Lp.W, ntiles, nt, kc, lane);
                        acc = __builtin_amdgcn_wmma_f32_16x16x32_bf16(
                                  false, a, false, bb, (short)0, acc, false, false);
                    }
#pragma unroll
                    for (int i = 0; i < 8; ++i) {
                        float v = acc[i] * sc + bi;
                        v = v > 0.0f ? v : 0.0f;
                        rmax[i] = fmaxf(rmax[i], v);
                    }
                }
                float m = rmax[0];
#pragma unroll
                for (int i = 1; i < 8; ++i) m = fmaxf(m, rmax[i]);
                m = fmaxf(m, __shfl_xor(m, 16, 32));  // combine lane halves (wave32)
                if (lane < 16)
                    outp[((size_t)(b * S + s)) * outStride + outOff + ncol] = m;
            }
        }
    }
}

// ---------------------------------------------------------------------------
// Host driver
// ---------------------------------------------------------------------------
extern "C" void kernel_launch(void* const* d_in, const int* in_sizes, int n_in,
                              void* d_out, int out_size, void* d_ws, size_t ws_size,
                              hipStream_t stream) {
    const int B = 16, N = 2048, S1 = 512, S2 = 128;

    // xyz may be first (dict insertion order) or last (alphabetical tree flatten)
    int xyzIdx, pstart;
    if (in_sizes[0] == B * 6 * N) { xyzIdx = 0; pstart = 1; }
    else                          { xyzIdx = n_in - 1; pstart = 0; }
    const float* xyz_in = (const float*)d_in[xyzIdx];

    // 21 layers: {cout, cin} in flatten order (sa1 b0..b2, sa2 b0..b2, sa3)
    static const int LDIM[21][2] = {
        {32, 6}, {32, 32}, {64, 32},   {64, 6}, {64, 64}, {128, 64},
        {64, 6}, {96, 64}, {128, 96},
        {64, 323}, {64, 64}, {128, 64}, {128, 323}, {128, 128}, {256, 128},
        {128, 323}, {128, 128}, {256, 128},
        {256, 643}, {512, 256}, {1024, 512}};

    char* ws = (char*)d_ws;
    size_t off = 0;
    auto alloc = [&](size_t bytes) -> char* {
        char* p = ws + off;
        off = (off + bytes + 255) & ~(size_t)255;
        return p;
    };

    float* pts = (float*)alloc((size_t)B * N * 3 * 4);
    float* nrm = (float*)alloc((size_t)B * N * 3 * 4);
    int*   fp1 = (int*)  alloc((size_t)B * S1 * 4);
    float* nx1 = (float*)alloc((size_t)B * S1 * 3 * 4);
    int*   g1a = (int*)  alloc((size_t)B * S1 * 16 * 4);
    int*   g1b = (int*)  alloc((size_t)B * S1 * 32 * 4);
    int*   g1c = (int*)  alloc((size_t)B * S1 * 128 * 4);
    float* l1  = (float*)alloc((size_t)B * S1 * 320 * 4);
    int*   fp2 = (int*)  alloc((size_t)B * S2 * 4);
    float* nx2 = (float*)alloc((size_t)B * S2 * 3 * 4);
    int*   g2a = (int*)  alloc((size_t)B * S2 * 32 * 4);
    int*   g2b = (int*)  alloc((size_t)B * S2 * 64 * 4);
    int*   g2c = (int*)  alloc((size_t)B * S2 * 128 * 4);
    float* l2  = (float*)alloc((size_t)B * S2 * 640 * 4);

    LayerP LP[21];
    for (int i = 0; i < 21; ++i) {
        int cout = LDIM[i][0], cin = LDIM[i][1], cinPad = (cin + 31) & ~31;
        unsigned short* wt = (unsigned short*)alloc((size_t)cinPad * cout * 2);
        LP[i].W = wt;
        LP[i].s = (const float*)d_in[pstart + 3 * i + 1];
        LP[i].b = (const float*)d_in[pstart + 3 * i + 2];
        LP[i].cin = cin; LP[i].cinPad = cinPad; LP[i].cout = cout;
        int tot = cinPad * cout;   // == (cinPad/32)*(cout/16)*512
        convw_kernel<<<(tot + 255) / 256, 256, 0, stream>>>(
            (const float*)d_in[pstart + 3 * i], wt, cin, cinPad, cout);
    }

    split_xyz_kernel<<<(B * N + 255) / 256, 256, 0, stream>>>(xyz_in, pts, nrm, B, N);

    fps_kernel<2048><<<B, FPS_THREADS, 0, stream>>>(pts, S1, fp1, nx1);
    ballquery_kernel<<<(B * S1 + 255) / 256, 256, 0, stream>>>(pts, N, nx1, S1, 16,  0.01f, g1a, B);
    ballquery_kernel<<<(B * S1 + 255) / 256, 256, 0, stream>>>(pts, N, nx1, S1, 32,  0.04f, g1b, B);
    ballquery_kernel<<<(B * S1 + 255) / 256, 256, 0, stream>>>(pts, N, nx1, S1, 128, 0.16f, g1c, B);

    auto launch_branch = [&](const float* coords, int Npts, const float* feats, int Cf,
                             const float* nxyz, const int* gidx, int S, int K, int mode,
                             float* outp, int ostr, int ooff, int li) {
        MlpP P; P.L[0] = LP[li]; P.L[1] = LP[li + 1]; P.L[2] = LP[li + 2];
        int c0p = LP[li].cinPad, c1 = LP[li].cout, c2 = LP[li + 1].cout;
        int aE = K * (c0p > c2 ? c0p : c2);
        size_t sh = (size_t)(aE + K * c1) * 2;
        sa_mlp_kernel<<<B * S, 128, sh, stream>>>(coords, Npts, feats, Cf, nxyz, gidx,
                                                  S, K, mode, outp, ostr, ooff, P, aE);
    };

    // SA1 branches -> l1 [B,512,320] at channel offsets 0,64,192
    launch_branch(pts, N, nrm, 3, nx1, g1a, S1, 16,  0, l1, 320, 0,   0);
    launch_branch(pts, N, nrm, 3, nx1, g1b, S1, 32,  0, l1, 320, 64,  3);
    launch_branch(pts, N, nrm, 3, nx1, g1c, S1, 128, 0, l1, 320, 192, 6);

    fps_kernel<512><<<B, FPS_THREADS, 0, stream>>>(nx1, S2, fp2, nx2);
    ballquery_kernel<<<(B * S2 + 255) / 256, 256, 0, stream>>>(nx1, S1, nx2, S2, 32,  0.04f, g2a, B);
    ballquery_kernel<<<(B * S2 + 255) / 256, 256, 0, stream>>>(nx1, S1, nx2, S2, 64,  0.16f, g2b, B);
    ballquery_kernel<<<(B * S2 + 255) / 256, 256, 0, stream>>>(nx1, S1, nx2, S2, 128, 0.64f, g2c, B);

    // SA2 branches -> l2 [B,128,640] at channel offsets 0,128,384
    launch_branch(nx1, S1, l1, 320, nx2, g2a, S2, 32,  0, l2, 640, 0,   9);
    launch_branch(nx1, S1, l1, 320, nx2, g2b, S2, 64,  0, l2, 640, 128, 12);
    launch_branch(nx1, S1, l1, 320, nx2, g2c, S2, 128, 0, l2, 640, 384, 15);

    // SA3 group_all -> d_out [B,1024]
    launch_branch(nx2, S2, l2, 640, nx2, g2c, 1, 128, 1, (float*)d_out, 1024, 0, 18);
}
